// CausalSelfAttention_82652350644696
// MI455X (gfx1250) — matrix operand order repaired and checked by reading, not compiled, not run on hardware
//
#include <hip/hip_runtime.h>

#define NB 4
#define SS 2048
#define DD 1024
#define NH 16
#define HD 64

typedef __attribute__((ext_vector_type(16))) _Float16 v16h;
typedef __attribute__((ext_vector_type(8)))  _Float16 v8h;
typedef __attribute__((ext_vector_type(4)))  _Float16 v4h;
typedef __attribute__((ext_vector_type(8)))  float    v8f;

static __device__ __forceinline__ v16h frag_join(v8h lo, v8h hi) {
  v16h o;
#pragma unroll
  for (int i = 0; i < 8; ++i) { o[i] = lo[i]; o[i + 8] = hi[i]; }
  return o;
}

// A fragment: 16(M) x 32(K), source row-major, leading dim ld (in halves).
// Lane<16: row=lane, K elems {k0..k0+7, k0+16..k0+23}; lane>=16: row=lane-16,
// K elems {k0+8..k0+15, k0+24..k0+31}.  (ISA 7.12.2, 16-bit A 16x32)
static __device__ __forceinline__ v16h load_a(const _Float16* p, int ld,
                                              int row0, int k0, int lane) {
  int hf = lane >> 4, r = lane & 15;
  const _Float16* base = p + (size_t)(row0 + r) * ld + k0 + hf * 8;
  return frag_join(*(const v8h*)base, *(const v8h*)(base + 16));
}

// B fragment: 32(K) x 16(N) with B[k][n] = W[n][k]; W row-major [ldw].
// Lane<16: col=lane, K = {k0..k0+7, k0+16..k0+23}; lane>=16: col=lane-16, K = {k0+8..k0+15, k0+24..k0+31}.
static __device__ __forceinline__ v16h load_b(const _Float16* w, int ldw,
                                              int n0, int k0, int lane) {
  int hf = lane >> 4, c = lane & 15;
  const _Float16* base = w + (size_t)(n0 + c) * ldw + k0 + hf * 8;
  return frag_join(*(const v8h*)base, *(const v8h*)(base + 16));
}

static __device__ __forceinline__ v8f wmma_f16(v16h a, v16h b, v8f c) {
  return __builtin_amdgcn_wmma_f32_16x16x32_f16(false, a, false, b,
                                                (short)0, c, false, false);
}

// ---------------------------------------------------------------- convert
__global__ __launch_bounds__(256) void f32_to_f16_kernel(const float* __restrict__ in,
                                                         _Float16* __restrict__ out,
                                                         int n4) {
  int i = blockIdx.x * blockDim.x + threadIdx.x;
  if (i < n4) {
    float4 f = ((const float4*)in)[i];
    v4h h;
    h[0] = (_Float16)f.x; h[1] = (_Float16)f.y;
    h[2] = (_Float16)f.z; h[3] = (_Float16)f.w;
    ((v4h*)out)[i] = h;
  }
}

// ------------------------------------------------- QKV gemm + fused RoPE
// qkv[g, e] = sum_d xh[g, d] * w_qkv[e, d], g in [0, B*S)
// Each wave: 16 rows x 64 cols (one head's full hd for q/k sections).
__global__ __launch_bounds__(256) void qkv_rope_kernel(
    const _Float16* __restrict__ xh, const _Float16* __restrict__ wh,
    _Float16* __restrict__ qh, _Float16* __restrict__ kh,
    _Float16* __restrict__ vTh) {
  int lane = threadIdx.x & 31;
  int gw = blockIdx.x * 8 + (threadIdx.x >> 5);   // 24576 wave tiles
  int tm = gw / 48, tn = gw % 48;
  int m0 = tm * 16, n0 = tn * 64;

  v8f acc[4] = {};
  for (int k0 = 0; k0 < DD; k0 += 32) {
    v16h a = load_a(xh, DD, m0, k0, lane);
#pragma unroll
    for (int t = 0; t < 4; ++t) {
      v16h b = load_b(wh, DD, n0 + t * 16, k0, lane);
      acc[t] = wmma_f16(a, b, acc[t]);
    }
  }

  int hf = lane >> 4, c = lane & 15;
  int sec = n0 / DD;             // 0=q 1=k 2=v
  int hh = (n0 % DD) / HD;       // head index
  int g0 = m0 + 8 * hf;          // element r -> global row g0+r
  int bi = g0 / SS;
  int s0 = g0 % SS;

  if (sec < 2) {
    // RoPE: pair (p, p+32) lives in (acc[t], acc[t+2]); cos/sin shared.
#pragma unroll
    for (int t = 0; t < 2; ++t) {
      int p = t * 16 + c;                           // 0..31
      float fr = expf(-(float)p * 0.2878231366f);   // 10000^(-p/32)
#pragma unroll
      for (int r = 0; r < 8; ++r) {
        float th = (float)(s0 + r) * fr;
        float cs = cosf(th), sn = sinf(th);
        float a0 = acc[t][r], a2 = acc[t + 2][r];
        acc[t][r]     = a0 * cs - a2 * sn;
        acc[t + 2][r] = a2 * cs + a0 * sn;
      }
    }
    _Float16* dst = (sec == 0) ? qh : kh;           // [B,H,S,hd]
#pragma unroll
    for (int t = 0; t < 4; ++t) {
      int p = t * 16 + c;
#pragma unroll
      for (int r = 0; r < 8; ++r)
        dst[(((size_t)(bi * NH + hh)) * SS + (s0 + r)) * HD + p] =
            (_Float16)acc[t][r];
    }
  } else {                                          // v, transposed [B,H,hd,S]
#pragma unroll
    for (int t = 0; t < 4; ++t) {
      int p = t * 16 + c;
#pragma unroll
      for (int r = 0; r < 8; ++r)
        vTh[(((size_t)(bi * NH + hh)) * HD + p) * SS + (s0 + r)] =
            (_Float16)acc[t][r];
    }
  }
}

// ---------------------------------------------- causal flash attention
// One wave: (b, h, 16-query tile). K-loop over 32-key steps.
__global__ __launch_bounds__(256) void attn_kernel(
    const _Float16* __restrict__ qh, const _Float16* __restrict__ kh,
    const _Float16* __restrict__ vTh, _Float16* __restrict__ aoh) {
  __shared__ _Float16 pbuf[8][16 * 32];             // per-wave P tile (1 KB)
  int lane = threadIdx.x & 31;
  int wv = threadIdx.x >> 5;
  int gw = blockIdx.x * 8 + wv;                     // 8192 wave tiles
  int qt = gw & 127;
  int hh = (gw >> 7) & 15;
  int bi = gw >> 11;
  int q0 = qt * 16;
  size_t bh = (size_t)bi * NH + hh;
  const _Float16* qp = qh + bh * SS * HD;
  const _Float16* kp = kh + bh * SS * HD;
  const _Float16* vp = vTh + bh * (size_t)HD * SS;

  v16h aQ0 = load_a(qp, HD, q0, 0, lane);
  v16h aQ1 = load_a(qp, HD, q0, 32, lane);

  v8f o[4] = {};
  float Mr[8], Lr[8];
#pragma unroll
  for (int r = 0; r < 8; ++r) { Mr[r] = -1e30f; Lr[r] = 0.0f; }

  int hf = lane >> 4, c = lane & 15;
  _Float16* pl = &pbuf[wv][0];
  int nkt = (q0 + 47) / 32;

  for (int kt = 0; kt < nkt; ++kt) {
    int k0 = kt * 32;
    if (kt + 1 < nkt)
      __builtin_prefetch(kp + (size_t)(k0 + 32) * HD, 0, 1);

    // scores 16x32: two 16-wide C tiles, K-dim = hd = 64 (2 wmma steps each)
    v8f s0 = {}, s1 = {};
    s0 = wmma_f16(aQ0, load_b(kp, HD, k0, 0, lane), s0);
    s0 = wmma_f16(aQ1, load_b(kp, HD, k0, 32, lane), s0);
    s1 = wmma_f16(aQ0, load_b(kp, HD, k0 + 16, 0, lane), s1);
    s1 = wmma_f16(aQ1, load_b(kp, HD, k0 + 16, 32, lane), s1);

    // scale + causal mask   (row m = r + 8*hf, col = c | c+16)
#pragma unroll
    for (int r = 0; r < 8; ++r) {
      int qg = q0 + 8 * hf + r;
      float v0 = s0[r] * 0.125f;
      float v1 = s1[r] * 0.125f;
      if (k0 + c > qg)      v0 = -1e30f;
      if (k0 + 16 + c > qg) v1 = -1e30f;
      s0[r] = v0; s1[r] = v1;
    }

    // online softmax; row reductions stay inside each 16-lane half
#pragma unroll
    for (int r = 0; r < 8; ++r) {
      float rm = fmaxf(s0[r], s1[r]);
      rm = fmaxf(rm, __shfl_xor(rm, 1, 32));
      rm = fmaxf(rm, __shfl_xor(rm, 2, 32));
      rm = fmaxf(rm, __shfl_xor(rm, 4, 32));
      rm = fmaxf(rm, __shfl_xor(rm, 8, 32));
      float nm = fmaxf(Mr[r], rm);
      float al = __expf(Mr[r] - nm);
      float p0 = __expf(s0[r] - nm);
      float p1 = __expf(s1[r] - nm);
      float rs = p0 + p1;
      rs += __shfl_xor(rs, 1, 32);
      rs += __shfl_xor(rs, 2, 32);
      rs += __shfl_xor(rs, 4, 32);
      rs += __shfl_xor(rs, 8, 32);
      Lr[r] = Lr[r] * al + rs;
      Mr[r] = nm;
      s0[r] = p0; s1[r] = p1;
#pragma unroll
      for (int t = 0; t < 4; ++t) o[t][r] *= al;
    }

    // re-fragment P (C layout -> A layout) through per-wave LDS tile
#pragma unroll
    for (int r = 0; r < 8; ++r) {
      int m = 8 * hf + r;
      pl[m * 32 + c]      = (_Float16)s0[r];
      pl[m * 32 + 16 + c] = (_Float16)s1[r];
    }
    asm volatile("s_wait_dscnt 0" ::: "memory");
    v16h aP = load_a(pl, 32, 0, 0, lane);
    asm volatile("s_wait_dscnt 0" ::: "memory");

    // O += P x V   (V^T rows are hd dims, contiguous over keys)
#pragma unroll
    for (int t = 0; t < 4; ++t) {
      v16h bv = load_b(vp + k0, SS, t * 16, 0, lane);
      o[t] = wmma_f16(aP, bv, o[t]);
    }
  }

  // normalize + store attention output f16 [B,S,D]
#pragma unroll
  for (int r = 0; r < 8; ++r) {
    float inv = 1.0f / Lr[r];
    int qg = q0 + 8 * hf + r;
#pragma unroll
    for (int t = 0; t < 4; ++t)
      aoh[((size_t)(bi * SS + qg)) * DD + hh * HD + t * 16 + c] =
          (_Float16)(o[t][r] * inv);
  }
}

// ----------------------------------------------------- output projection
__global__ __launch_bounds__(256) void out_proj_kernel(
    const _Float16* __restrict__ aoh, const _Float16* __restrict__ wh,
    float* __restrict__ out) {
  int lane = threadIdx.x & 31;
  int gw = blockIdx.x * 8 + (threadIdx.x >> 5);    // 8192 wave tiles
  int tm = gw / 16, tn = gw % 16;
  int m0 = tm * 16, n0 = tn * 64;

  v8f acc[4] = {};
  for (int k0 = 0; k0 < DD; k0 += 32) {
    v16h a = load_a(aoh, DD, m0, k0, lane);
#pragma unroll
    for (int t = 0; t < 4; ++t) {
      v16h b = load_b(wh, DD, n0 + t * 16, k0, lane);
      acc[t] = wmma_f16(a, b, acc[t]);
    }
  }
  int hf = lane >> 4, c = lane & 15;
#pragma unroll
  for (int t = 0; t < 4; ++t)
#pragma unroll
    for (int r = 0; r < 8; ++r)
      out[(size_t)(m0 + 8 * hf + r) * DD + n0 + t * 16 + c] = acc[t][r];
}

// ---------------------------------------------------------------- launch
extern "C" void kernel_launch(void* const* d_in, const int* in_sizes, int n_in,
                              void* d_out, int out_size, void* d_ws, size_t ws_size,
                              hipStream_t stream) {
  const float* x     = (const float*)d_in[0];   // [4,2048,1024]
  const float* w_qkv = (const float*)d_in[1];   // [3072,1024]
  const float* w_out = (const float*)d_in[2];   // [1024,1024]
  float* outp = (float*)d_out;

  const size_t N_X  = (size_t)NB * SS * DD;     // 8388608
  const size_t N_WQ = (size_t)3 * DD * DD;      // 3145728
  const size_t N_WO = (size_t)DD * DD;          // 1048576

  _Float16* ws = (_Float16*)d_ws;
  _Float16* xh    = ws;
  _Float16* wqkvh = xh + N_X;
  _Float16* wouth = wqkvh + N_WQ;
  _Float16* qhp   = wouth + N_WO;
  _Float16* khp   = qhp + N_X;
  _Float16* vTp   = khp + N_X;
  _Float16* aop   = vTp + N_X;

  // fp32 -> f16 conversions
  f32_to_f16_kernel<<<(int)((N_X / 4 + 255) / 256), 256, 0, stream>>>(x, xh, (int)(N_X / 4));
  f32_to_f16_kernel<<<(int)((N_WQ / 4 + 255) / 256), 256, 0, stream>>>(w_qkv, wqkvh, (int)(N_WQ / 4));
  f32_to_f16_kernel<<<(int)((N_WO / 4 + 255) / 256), 256, 0, stream>>>(w_out, wouth, (int)(N_WO / 4));

  // QKV projection + RoPE: 512 x 48 wave tiles / 8 waves per block
  qkv_rope_kernel<<<3072, 256, 0, stream>>>(xh, wqkvh, qhp, khp, vTp);

  // flash attention: 4*16*128 = 8192 wave tiles
  attn_kernel<<<1024, 256, 0, stream>>>(qhp, khp, vTp, aop);

  // output projection: 512 x 16 wave tiles
  out_proj_kernel<<<1024, 256, 0, stream>>>(aop, wouth, outp);
}
